// IHRLayer_28982439313610
// MI455X (gfx1250) — matrix operand order repaired and checked by reading, not compiled, not run on hardware
//
#include <hip/hip_runtime.h>
#include <stdint.h>

typedef __attribute__((ext_vector_type(16))) __bf16 v16bf;
typedef __attribute__((ext_vector_type(8)))  float  v8f;

#define QN    16384      // BEV queries = 128*128 pixels
#define CCH   256        // channels
#define NCAM  6
#define HFEAT 32
#define WFEAT 88
#define DDEP  64
#define KTOT  (9*CCH)    // implicit-GEMM K = 2304

// ---------- helpers ----------
static __device__ __forceinline__ __bf16 f2bf(float f) {
  union { float f; uint32_t u; } a; a.f = f;
  uint32_t u = a.u;
  uint32_t r = (u + 0x7FFFu + ((u >> 16) & 1u)) >> 16;   // RNE
  union { uint16_t s; __bf16 b; } o; o.s = (uint16_t)r;
  return o.b;
}
static __device__ __forceinline__ float sigm(float x) { return 1.f / (1.f + __expf(-x)); }

// ---------- kernel 0a: zero the A-fragment zero page (re-done every call) ----------
__global__ void k_zero(uint32_t* __restrict__ zp) {
  zp[threadIdx.x] = 0u;          // 256 * 4 B = 1 KB
}

// ---------- kernel 0b: repack conv weights (OIHW f32) -> bf16 [n][k = kpos*256+c] ----------
__global__ void k_repack(const float* __restrict__ conv_w, __bf16* __restrict__ wb) {
  const int k = blockIdx.x;          // 0..2303
  const int n = threadIdx.x;         // 0..255
  const int kpos = k >> 8, c = k & 255;
  wb[n * KTOT + k] = f2bf(conv_w[n * KTOT + c * 9 + kpos]);
}

// ---------- kernel 1: per-query heads + projection + gather + camera-weighted sum ----------
// one block (256 thr) per query; writes channels-last bf16 activations + rp output
__global__ __launch_bounds__(256) void k_sample(
    const float* __restrict__ feat, const float* __restrict__ bevq,
    const float* __restrict__ bevp, const float* __restrict__ refpts,
    const float* __restrict__ pcr,  const float* __restrict__ l2i,
    const float* __restrict__ dprob,
    const float* __restrict__ w_pos, const float* __restrict__ b_pos,
    const float* __restrict__ w_img, const float* __restrict__ b_img,
    const float* __restrict__ w_att, const float* __restrict__ b_att,
    __bf16* __restrict__ xcl, float* __restrict__ rp_out)
{
  __shared__ float qs[CCH];
  __shared__ float dots[19];
  __shared__ float rp_s[3];
  __shared__ float camP[NCAM][5];    // wgt, x0, y0, wx, wy

  const int q = blockIdx.x;
  const int c = threadIdx.x;

  qs[c] = bevq[q * CCH + c] + bevp[q * CCH + c];
  __syncthreads();

  if (c < 19) {                       // 19 dot products of length 256
    float s = 0.f;
    if (c == 0) {
      for (int i = 0; i < CCH; ++i) s += qs[i] * w_pos[i];
      s += b_pos[0];
    } else if (c < 13) {
      const int j = c - 1;
      for (int i = 0; i < CCH; ++i) s += qs[i] * w_img[i * 12 + j];
      s += b_img[j];
    } else {
      const int j = c - 13;
      for (int i = 0; i < CCH; ++i) s += qs[i] * w_att[i * 6 + j];
      s += b_att[j];
    }
    dots[c] = s;
  }
  __syncthreads();

  if (c == 0) {                       // reference point z-refinement -> rp output
    const float rx = refpts[q * 3 + 0], ry = refpts[q * 3 + 1], rz = refpts[q * 3 + 2];
    float x = fminf(fmaxf(rz, 0.f), 1.f);
    const float is = __logf(fmaxf(x, 1e-5f) / fmaxf(1.f - x, 1e-5f));
    const float z = sigm(is + dots[0]);
    rp_s[0] = rx; rp_s[1] = ry; rp_s[2] = z;
    rp_out[q * 3 + 0] = rx; rp_out[q * 3 + 1] = ry; rp_out[q * 3 + 2] = z;
  }
  __syncthreads();

  if (c < NCAM) {                     // per-camera projection + depth weight
    const int n = c;
    const float px = rp_s[0] * (pcr[3] - pcr[0]) + pcr[0];
    const float py = rp_s[1] * (pcr[4] - pcr[1]) + pcr[1];
    const float pz = rp_s[2] * (pcr[5] - pcr[2]) + pcr[2];
    const float* L = l2i + n * 16;
    const float cx = L[0] * px + L[1] * py + L[2]  * pz + L[3];
    const float cy = L[4] * px + L[5] * py + L[6]  * pz + L[7];
    const float cz = L[8] * px + L[9] * py + L[10] * pz + L[11];
    const float zc = fmaxf(cz, 1e-5f);
    const float xn = (cx / zc / 704.f - 0.5f) * 2.f;
    const float yn = (cy / zc / 256.f - 0.5f) * 2.f;
    const float rdn = (cz / (float)DDEP - 0.5f) * 2.f;
    const bool msk = (cz > 1e-5f) && (xn > -1.f) && (xn < 1.f) &&
                     (yn > -1.f) && (yn < 1.f) && (rdn > -1.f) && (rdn < 1.f);
    const float xs = xn + dots[1 + 2 * n] * (1.f / (float)WFEAT);
    const float ys = yn + dots[2 + 2 * n] * (1.f / (float)HFEAT);
    // bilinear coords in feature map
    const float gx = (xs + 1.f) * 0.5f * (float)WFEAT - 0.5f;
    const float gy = (ys + 1.f) * 0.5f * (float)HFEAT - 0.5f;
    const float fx0 = floorf(gx), fy0 = floorf(gy);
    const float wx = gx - fx0, wy = gy - fy0;
    // trilinear depth probability
    const float gz = (rdn + 1.f) * 0.5f * (float)DDEP - 0.5f;
    const float fz0 = floorf(gz);
    const float wz = gz - fz0;
    const int x0 = (int)fx0, y0 = (int)fy0, z0 = (int)fz0;
    const float* dpn = dprob + (size_t)n * DDEP * HFEAT * WFEAT;
    float dp = 0.f;
    #pragma unroll
    for (int dzi = 0; dzi < 2; ++dzi)
      #pragma unroll
      for (int dyi = 0; dyi < 2; ++dyi)
        #pragma unroll
        for (int dxi = 0; dxi < 2; ++dxi) {
          const int xi = x0 + dxi, yi = y0 + dyi, zi = z0 + dzi;
          if (xi >= 0 && xi < WFEAT && yi >= 0 && yi < HFEAT && zi >= 0 && zi < DDEP) {
            const float wv = (dxi ? wx : 1.f - wx) * (dyi ? wy : 1.f - wy) * (dzi ? wz : 1.f - wz);
            dp += wv * dpn[(zi * HFEAT + yi) * WFEAT + xi];
          }
        }
    const float attn_s = sigm(dots[13 + n]);
    camP[n][0] = msk ? attn_s * dp : 0.f;
    camP[n][1] = (float)x0; camP[n][2] = (float)y0;
    camP[n][3] = wx;        camP[n][4] = wy;
  }
  __syncthreads();

  // per-channel gather + camera-weighted accumulate
  float acc = 0.f;
  for (int n = 0; n < NCAM; ++n) {
    const float wgt = camP[n][0];
    if (wgt != 0.f) {
      const int x0 = (int)camP[n][1], y0 = (int)camP[n][2];
      const float wx = camP[n][3], wy = camP[n][4];
      const float* fb = feat + (size_t)(n * CCH + c) * (HFEAT * WFEAT);
      const bool vx0 = (x0 >= 0) && (x0 < WFEAT);
      const bool vx1 = (x0 + 1 >= 0) && (x0 + 1 < WFEAT);
      const bool vy0 = (y0 >= 0) && (y0 < HFEAT);
      const bool vy1 = (y0 + 1 >= 0) && (y0 + 1 < HFEAT);
      const float v00 = (vx0 && vy0) ? fb[y0 * WFEAT + x0]           : 0.f;
      const float v10 = (vx1 && vy0) ? fb[y0 * WFEAT + x0 + 1]       : 0.f;
      const float v01 = (vx0 && vy1) ? fb[(y0 + 1) * WFEAT + x0]     : 0.f;
      const float v11 = (vx1 && vy1) ? fb[(y0 + 1) * WFEAT + x0 + 1] : 0.f;
      acc += wgt * (v00 * (1.f - wx) * (1.f - wy) + v10 * wx * (1.f - wy) +
                    v01 * (1.f - wx) * wy         + v11 * wx * wy);
    }
  }
  xcl[(size_t)q * CCH + c] = f2bf(acc);   // channels-last bf16 for the WMMA conv
}

// ---------- kernel 2: 3x3 conv 256->256 as implicit GEMM on v_wmma_f32_16x16x32_bf16 ----------
// one wave = 16-pixel x 16-out-channel tile; K-loop of 9 taps x 8 chunks of 32 channels.
// Halo pixels outside the image read from a zeroed scratch page -> branch-free, uniform
// control flow (EXEC stays all-ones around every WMMA), loads clause as b128 pairs.
__global__ __launch_bounds__(256) void k_conv(
    const __bf16* __restrict__ xcl, const __bf16* __restrict__ wb,
    const __bf16* __restrict__ zp,
    const float* __restrict__ conv_b,
    const float* __restrict__ bn_g, const float* __restrict__ bn_b,
    const float* __restrict__ bn_m, const float* __restrict__ bn_v,
    float* __restrict__ y)
{
  const int lane   = threadIdx.x & 31;
  const int gw     = blockIdx.x * 8 + (threadIdx.x >> 5);
  const int tile_m = gw >> 4;                  // 0..1023 pixel strips
  const int tile_n = gw & 15;                  // 0..15 out-channel tiles
  const int pbase  = tile_m << 4;              // 16 consecutive pixels in one row
  const int h      = pbase >> 7;
  const int w      = (pbase & 127) + (lane & 15);   // A-row M = lane&15
  const int kb     = (lane >> 4) << 3;         // A layout: lanes 16-31 hold K offset +8

  union ABf { v16bf v; uint4 q[2]; };
  v8f acc = {};

  #pragma unroll
  for (int kpos = 0; kpos < 9; ++kpos) {
    const int dy = kpos / 3 - 1, dx = kpos % 3 - 1;
    const int hh = h + dy, ww = w + dx;
    const bool pv = (hh >= 0) && (hh < 128) && (ww >= 0) && (ww < 128);
    // branch-free halo: out-of-image lanes read zeros from the zero page
    const __bf16* abase = pv ? (xcl + ((size_t)((hh << 7) + ww) * CCH + kb)) : (zp + kb);
    const __bf16* bbase = wb + ((size_t)((tile_n << 4) + (lane & 15)) * KTOT + kpos * CCH + kb);
    if (kpos < 8)
      __builtin_prefetch((const void*)(bbase + (size_t)CCH), 0, 1);
    #pragma unroll
    for (int cb = 0; cb < CCH; cb += 32) {
      ABf A;   // elements 0-7: K=kb..kb+7 ; elements 8-15: K=16+kb..16+kb+7
      A.q[0] = *(const uint4*)(abase + cb);
      A.q[1] = *(const uint4*)(abase + cb + 16);
      ABf Bv;  // B layout: lane = N, elements same K order as A
      Bv.q[0] = *(const uint4*)(bbase + cb);
      Bv.q[1] = *(const uint4*)(bbase + cb + 16);
      acc = __builtin_amdgcn_wmma_f32_16x16x32_bf16(false, A.v, false, Bv.v,
                                                    (short)0, acc, false, false);
    }
  }

  // epilogue: bias + BN + ReLU, D layout: lane = N, VGPR r = M (lanes>=16: M+8)
  const int nglob = (tile_n << 4) + (lane & 15);
  const float sc = bn_g[nglob] * rsqrtf(bn_v[nglob] + 1e-5f);
  const float sh = bn_b[nglob] - bn_m[nglob] * sc;
  const float cbv = conv_b[nglob];
  const int mh = (lane >> 4) << 3;
  #pragma unroll
  for (int r = 0; r < 8; ++r) {
    float v = (acc[r] + cbv) * sc + sh;
    v = fmaxf(v, 0.f);
    y[(size_t)(pbase + mh + r) * CCH + nglob] = v;   // pixel-major f32
  }
}

// ---------- kernel 3a/3b: SE global pool + 256x256 GEMV + sigmoid ----------
__global__ void k_se_partial(const float* __restrict__ y, float* __restrict__ partial) {
  const int b = blockIdx.x, c = threadIdx.x;
  float s = 0.f;
  const int p0 = b * 128;
  for (int i = 0; i < 128; ++i) s += y[(size_t)(p0 + i) * CCH + c];
  partial[b * CCH + c] = s;
}
__global__ void k_se_fc(const float* __restrict__ partial, const float* __restrict__ se_w,
                        const float* __restrict__ se_b, float* __restrict__ se_sig) {
  __shared__ float s[CCH];
  const int c = threadIdx.x;
  float t = 0.f;
  for (int b = 0; b < 128; ++b) t += partial[b * CCH + c];
  s[c] = t * (1.f / (float)QN);
  __syncthreads();
  float d = se_b[c];
  for (int i = 0; i < CCH; ++i) d += s[i] * se_w[c * CCH + i];
  se_sig[c] = sigm(d);
}

// ---------- kernel 4a: per-pixel channel max/mean of SE-scaled activations ----------
__global__ __launch_bounds__(256) void k_chanred(
    const float* __restrict__ y, const float* __restrict__ se_sig,
    float* __restrict__ xmax, float* __restrict__ xmean) {
  __shared__ float smax[CCH], ssum[CCH];
  const int p = blockIdx.x, c = threadIdx.x;
  const float v = y[(size_t)p * CCH + c] * se_sig[c];
  smax[c] = v; ssum[c] = v;
  __syncthreads();
  for (int s = 128; s > 0; s >>= 1) {
    if (c < s) { smax[c] = fmaxf(smax[c], smax[c + s]); ssum[c] += ssum[c + s]; }
    __syncthreads();
  }
  if (c == 0) { xmax[p] = smax[0]; xmean[p] = ssum[0] * (1.f / (float)CCH); }
}

// ---------- kernel 4b: CBAM 7x7 spatial conv + sigmoid ----------
__global__ void k_cbam(const float* __restrict__ xmax, const float* __restrict__ xmean,
                       const float* __restrict__ cw, const float* __restrict__ cb,
                       float* __restrict__ amap) {
  const int p = blockIdx.x * blockDim.x + threadIdx.x;
  if (p >= QN) return;
  const int h = p >> 7, w = p & 127;
  float a = cb[0];
  for (int ky = 0; ky < 7; ++ky) {
    const int hh = h + ky - 3;
    if (hh < 0 || hh >= 128) continue;
    for (int kx = 0; kx < 7; ++kx) {
      const int ww = w + kx - 3;
      if (ww < 0 || ww >= 128) continue;
      const int pp = hh * 128 + ww;
      a += cw[ky * 7 + kx] * xmax[pp] + cw[49 + ky * 7 + kx] * xmean[pp];
    }
  }
  amap[p] = sigm(a);
}

// ---------- kernel 4c: final scaling -> out[p][c] ----------
__global__ void k_final(const float* __restrict__ y, const float* __restrict__ se_sig,
                        const float* __restrict__ amap, float* __restrict__ out) {
  const int p = blockIdx.x, c = threadIdx.x;
  out[(size_t)p * CCH + c] = y[(size_t)p * CCH + c] * se_sig[c] * amap[p];
}

// ---------- host ----------
extern "C" void kernel_launch(void* const* d_in, const int* in_sizes, int n_in,
                              void* d_out, int out_size, void* d_ws, size_t ws_size,
                              hipStream_t stream) {
  const float* feat    = (const float*)d_in[0];
  const float* bevq    = (const float*)d_in[1];
  const float* bevp    = (const float*)d_in[2];
  const float* refpts  = (const float*)d_in[3];
  const float* pcr     = (const float*)d_in[4];
  const float* l2i     = (const float*)d_in[5];
  const float* dprob   = (const float*)d_in[6];
  const float* w_pos   = (const float*)d_in[7];
  const float* b_pos   = (const float*)d_in[8];
  const float* w_img   = (const float*)d_in[9];
  const float* b_img   = (const float*)d_in[10];
  const float* w_att   = (const float*)d_in[11];
  const float* b_att   = (const float*)d_in[12];
  const float* conv_w  = (const float*)d_in[13];
  const float* conv_b  = (const float*)d_in[14];
  const float* bn_g    = (const float*)d_in[15];
  const float* bn_b    = (const float*)d_in[16];
  const float* bn_m    = (const float*)d_in[17];
  const float* bn_v    = (const float*)d_in[18];
  const float* se_w    = (const float*)d_in[19];
  const float* se_b    = (const float*)d_in[20];
  const float* cbam_w  = (const float*)d_in[21];
  const float* cbam_b  = (const float*)d_in[22];

  float* out    = (float*)d_out;
  float* rp_out = out + (size_t)QN * CCH;       // tuple: (out, rp) concatenated

  // workspace carve-up (~25.5 MB total)
  char* ws = (char*)d_ws;
  __bf16* xcl    = (__bf16*)(ws + 0);                       //  8,388,608 B
  __bf16* wb     = (__bf16*)(ws + 8388608);                 //  1,179,648 B
  float*  y      = (float*) (ws + 9568256);                 // 16,777,216 B
  float*  part   = (float*) (ws + 26345472);                //    131,072 B
  float*  se_sig = (float*) (ws + 26476544);                //      1,024 B
  float*  xmax   = (float*) (ws + 26477568);                //     65,536 B
  float*  xmean  = (float*) (ws + 26543104);                //     65,536 B
  float*  amap   = (float*) (ws + 26608640);                //     65,536 B
  __bf16* zpage  = (__bf16*)(ws + 26674176);                //      1,024 B (zeroed)

  k_zero<<<1, 256, 0, stream>>>((uint32_t*)zpage);
  k_repack<<<KTOT, CCH, 0, stream>>>(conv_w, wb);
  k_sample<<<QN, CCH, 0, stream>>>(feat, bevq, bevp, refpts, pcr, l2i, dprob,
                                   w_pos, b_pos, w_img, b_img, w_att, b_att,
                                   xcl, rp_out);
  k_conv<<<QN * 16 / (16 * 8), 256, 0, stream>>>(xcl, wb, zpage, conv_b,
                                                 bn_g, bn_b, bn_m, bn_v, y); // 2048 blocks
  k_se_partial<<<128, CCH, 0, stream>>>(y, part);
  k_se_fc<<<1, CCH, 0, stream>>>(part, se_w, se_b, se_sig);
  k_chanred<<<QN, CCH, 0, stream>>>(y, se_sig, xmax, xmean);
  k_cbam<<<QN / 256, 256, 0, stream>>>(xmax, xmean, cbam_w, cbam_b, amap);
  k_final<<<QN, CCH, 0, stream>>>(y, se_sig, amap, out);
}
